// ShapeConstraint_5720896438364
// MI455X (gfx1250) — matrix operand order repaired and checked by reading, not compiled, not run on hardware
//
#include <hip/hip_runtime.h>
#include <hip/hip_bf16.h>

// ---------------------------------------------------------------------------
// Problem constants (from reference): output/target [8192,1,48,48] f32,
// psu [6,9,48,48] f32, mu [6,9] f32, scalar f32 loss out.
// ---------------------------------------------------------------------------
#define NB      8192        // batch
#define HW      2304        // 48*48 pixels per image
#define NIJ     54          // 6*9 shear filters
#define BLK1    256         // stage-1 workgroups
#define IPB     32          // images per stage-1 block (256*32 = 8192)
#define T1      192         // stage-1 threads: 192 thr * 3 float4 = 2304 pixels
#define T2      128         // stage-2 threads (4 waves)

typedef __attribute__((ext_vector_type(16))) _Float16 v16h;
typedef __attribute__((ext_vector_type(8)))  float    v8f;

// ---------------------------------------------------------------------------
// Stage 1: streaming reduction over output/target (HBM-bound: 151 MB reads).
//   partial[blk*2304 + p] = sum over this block's 32 images of |diff| at p
//   psq[blk]              = sum over this block's 32 images of diff^2
// Coalesced float4 loads; no atomics -> bit-deterministic across replays.
// ---------------------------------------------------------------------------
__global__ __launch_bounds__(T1) void sc_partial_kernel(
    const float* __restrict__ out, const float* __restrict__ tgt,
    float* __restrict__ partial, float* __restrict__ psq)
{
    const int t   = threadIdx.x;          // 0..191
    const int blk = blockIdx.x;           // 0..255
    const int b0  = blk * IPB;

    float4 acc0 = {0.f,0.f,0.f,0.f};
    float4 acc1 = {0.f,0.f,0.f,0.f};
    float4 acc2 = {0.f,0.f,0.f,0.f};
    float  sq   = 0.f;

    const int p0 = 4 * t;                 // float4 slot k: pixel base p0 + k*768
    for (int i = 0; i < IPB; ++i) {
        const int base = (b0 + i) * HW;
        #pragma unroll
        for (int k = 0; k < 3; ++k) {
            const int idx = base + p0 + k * (4 * T1);
            const float4 o = *(const float4*)(out + idx);
            const float4 g = *(const float4*)(tgt + idx);
            const float dx = o.x - g.x, dy = o.y - g.y,
                        dz = o.z - g.z, dw = o.w - g.w;
            sq += dx*dx + dy*dy + dz*dz + dw*dw;
            float4* a = (k == 0) ? &acc0 : (k == 1) ? &acc1 : &acc2;
            a->x += fabsf(dx); a->y += fabsf(dy);
            a->z += fabsf(dz); a->w += fabsf(dw);
        }
    }

    float* pb = partial + blk * HW;
    *(float4*)(pb + p0)            = acc0;
    *(float4*)(pb + p0 + 4 * T1)   = acc1;
    *(float4*)(pb + p0 + 8 * T1)   = acc2;

    // block-reduce sq (fixed order -> deterministic)
    __shared__ float s[T1];
    s[t] = sq;
    __syncthreads();
    if (t < 64) s[t] += s[t + 64] + s[t + 128];
    __syncthreads();
    if (t == 0) {
        float v = 0.f;
        for (int i = 0; i < 64; ++i) v += s[i];
        psq[blk] = v;
    }
}

// helpers: fill 8 consecutive f16 fragment halves from two float4s
#define ABS8(F, base, q0, q1)                       \
    F[base+0] = (_Float16)fabsf(q0.x);              \
    F[base+1] = (_Float16)fabsf(q0.y);              \
    F[base+2] = (_Float16)fabsf(q0.z);              \
    F[base+3] = (_Float16)fabsf(q0.w);              \
    F[base+4] = (_Float16)fabsf(q1.x);              \
    F[base+5] = (_Float16)fabsf(q1.y);              \
    F[base+6] = (_Float16)fabsf(q1.z);              \
    F[base+7] = (_Float16)fabsf(q1.w);

#define PUT8(F, base, q0, q1)                       \
    F[base+0] = (_Float16)q0.x;                     \
    F[base+1] = (_Float16)q0.y;                     \
    F[base+2] = (_Float16)q0.z;                     \
    F[base+3] = (_Float16)q0.w;                     \
    F[base+4] = (_Float16)q1.x;                     \
    F[base+5] = (_Float16)q1.y;                     \
    F[base+6] = (_Float16)q1.z;                     \
    F[base+7] = (_Float16)q1.w;

// ---------------------------------------------------------------------------
// Stage 2: finalize. One workgroup, 4 waves.
//   Phase 1: A[p] = (1/8192) * sum_blk partial[blk][p]  (float4 sweeps -> LDS)
//   Phase 2: l1[ij] = sum_p |psu[ij,p]| * A[p] via v_wmma_f32_16x16x32_f16.
//     A-fragment layout (16-bit A 16x32, ISA 7.12.2) in AMD table order per lane:
//       halves 0..7  -> row[kb + hi*8 + 0..7]        (2x global_load_b128)
//       halves 8..15 -> row[kb + 16 + hi*8 + 0..7]   (2x global_load_b128)
//     B-fragment (32x16): halves 0..7 -> A[kb + hi*8 + 0..7], 8..15 -> A[kb + 16 + hi*8 + 0..7] (4x ds_load_b128)
//     Rows ij>=54 use a clamped pointer; their D rows are never read, so no
//     masking / EXEC divergence is needed anywhere.
//   Phase 3: loss = sqtot/(B*HW) + 0.5 * sum mu_ij * (l1_ij/HW)^2.
// ---------------------------------------------------------------------------
__global__ __launch_bounds__(T2) void sc_final_kernel(
    const float* __restrict__ psu, const float* __restrict__ mu,
    const float* __restrict__ partial, const float* __restrict__ psq,
    float* __restrict__ loss_out)
{
    __shared__ __align__(16) float A[HW];   // 9216 B
    __shared__ float l1s[64];
    __shared__ float sqtot;

    const int t = threadIdx.x;     // 0..127

    // ---- Phase 1: fold 256 per-block partials (fixed order, float4) ----
    const float invNB = 1.0f / (float)NB;
    for (int f = t; f < HW / 4; f += T2) {
        float4 s = {0.f,0.f,0.f,0.f};
        for (int b = 0; b < BLK1; ++b) {
            const float4 v = *(const float4*)(partial + b * HW + 4 * f);
            s.x += v.x; s.y += v.y; s.z += v.z; s.w += v.w;
        }
        float4 r = {s.x * invNB, s.y * invNB, s.z * invNB, s.w * invNB};
        *(float4*)(&A[4 * f]) = r;
    }
    if (t == 0) {
        float s = 0.f;
        for (int b = 0; b < BLK1; ++b) s += psq[b];
        sqtot = s;
    }
    __syncthreads();

    // ---- Phase 2: WMMA matvec, D = |psu|(16-row tile) x A_rep(2304x16) ----
    const int wave = t >> 5;             // 0..3 -> ij rows [wave*16, wave*16+16)
    const int lane = t & 31;
    const int m    = lane & 15;          // A-fragment row within tile
    const int hi   = lane >> 4;          // lane half-group
    const int ij   = wave * 16 + m;
    const float* prow = psu + (ij < NIJ ? ij : 0) * HW;  // clamp: rows >=54 unused
    const float* pA   = prow + hi * 8;        // halves 0..7 source
    const float* pB   = prow + 16 + hi * 8;   // halves 8..15 source

    v8f c = {0.f,0.f,0.f,0.f,0.f,0.f,0.f,0.f};

    for (int kb = 0; kb < HW; kb += 32) {
        const float4 q0 = *(const float4*)(pA + kb);
        const float4 q1 = *(const float4*)(pA + kb + 4);
        const float4 q2 = *(const float4*)(pB + kb);
        const float4 q3 = *(const float4*)(pB + kb + 4);
        const float4 r0 = *(const float4*)(&A[kb + hi * 8]);
        const float4 r1 = *(const float4*)(&A[kb + hi * 8 + 4]);
        const float4 r2 = *(const float4*)(&A[kb + 16 + hi * 8]);
        const float4 r3 = *(const float4*)(&A[kb + 16 + hi * 8 + 4]);

        v16h a, b;
        ABS8(a, 0, q0, q1)
        ABS8(a, 8, q2, q3)
        PUT8(b, 0, r0, r1)
        PUT8(b, 8, r2, r3)

        c = __builtin_amdgcn_wmma_f32_16x16x32_f16(
                /*neg_a=*/false, a, /*neg_b=*/false, b,
                /*c_mod=*/(short)0, c, /*reuse_a=*/false, /*reuse_b=*/false);
    }

    // f32 C/D layout: lane 0 holds (M=0..7, N=0), lane 16 holds (M=8..15, N=0).
    if ((lane & 15) == 0) {
        #pragma unroll
        for (int v = 0; v < 8; ++v)
            l1s[wave * 16 + hi * 8 + v] = c[v];
    }
    __syncthreads();

    // ---- Phase 3: assemble scalar loss ----
    if (t == 0) {
        const float invHW = 1.0f / (float)HW;
        float loss = sqtot / ((float)NB * (float)HW);   // MSE term
        float acc = 0.f;
        for (int r = 0; r < NIJ; ++r) {
            const float v = l1s[r] * invHW;             // l1[ij]
            acc += mu[r] * v * v;
        }
        loss_out[0] = loss + 0.5f * acc;                // GAMMA = 1
    }
}

// ---------------------------------------------------------------------------
extern "C" void kernel_launch(void* const* d_in, const int* in_sizes, int n_in,
                              void* d_out, int out_size, void* d_ws, size_t ws_size,
                              hipStream_t stream) {
    const float* out_t = (const float*)d_in[0];   // output [8192,1,48,48]
    const float* tgt   = (const float*)d_in[1];   // target [8192,1,48,48]
    const float* psu   = (const float*)d_in[2];   // psu    [6,9,48,48]
    const float* mu    = (const float*)d_in[3];   // mu     [6,9]
    float*       loss  = (float*)d_out;           // scalar

    float* partial = (float*)d_ws;                // BLK1*HW floats
    float* psq     = partial + (size_t)BLK1 * HW; // BLK1 floats

    sc_partial_kernel<<<BLK1, T1, 0, stream>>>(out_t, tgt, partial, psq);
    sc_final_kernel<<<1, T2, 0, stream>>>(psu, mu, partial, psq, loss);
}
